// Model_39676907885771
// MI455X (gfx1250) — matrix operand order repaired and checked by reading, not compiled, not run on hardware
//
#include <hip/hip_runtime.h>
#include <hip/hip_bf16.h>
#include <math.h>

typedef __attribute__((ext_vector_type(16))) __bf16 v16bf;
typedef __attribute__((ext_vector_type(8)))  float  v8f;

#define SCALE_QK 0.125f   // 1/sqrt(64)

__device__ __forceinline__ v8f wmma_bf16(v16bf a, v16bf b, v8f c) {
  // D = A(16x32 bf16) x B(32x16 bf16) + C(16x16 f32)
  return __builtin_amdgcn_wmma_f32_16x16x32_bf16(
      /*neg_a=*/false, a, /*neg_b=*/false, b,
      /*c_mod=*/(short)0, c, /*reuse_a=*/false, /*reuse_b=*/false);
}

// Generic A-fragment loader (f32 source -> bf16 frag), arbitrary strides.
// A is 16x32 (MxK). Lane L<16: row=row_base+L, elems 0..7 = K{0..7}, 8..15 = K{16..23};
// lanes 16..31 shift K by +8 within each 16-wide half (per ISA 16-bit A layout).
__device__ __forceinline__ v16bf load_a_frag(const float* __restrict__ base,
                                             int stride_row, int stride_k,
                                             int row_base, int k_base, int lane) {
  const int row  = row_base + (lane & 15);
  const int koff = ((lane >> 4) & 1) * 8;
  v16bf a;
#pragma unroll
  for (int e = 0; e < 8; ++e)
    a[e] = (__bf16)base[row * stride_row + (k_base + koff + e) * stride_k];
#pragma unroll
  for (int e = 0; e < 8; ++e)
    a[8 + e] = (__bf16)base[row * stride_row + (k_base + 16 + koff + e) * stride_k];
  return a;
}

// Generic B-fragment loader. B is 32x16 (KxN). Lane L holds column N = L&15;
// elems 0..7 = K{0..7}, 8..15 = K{16..23}; lanes 16..31 shift K by +8 (matches A).
__device__ __forceinline__ v16bf load_b_frag(const float* __restrict__ base,
                                             int stride_n, int stride_k,
                                             int n_base, int k_base, int lane) {
  const int n    = n_base + (lane & 15);
  const int koff = ((lane >> 4) & 1) * 8;
  v16bf b;
#pragma unroll
  for (int e = 0; e < 8; ++e)
    b[e] = (__bf16)base[n * stride_n + (k_base + koff + e) * stride_k];
#pragma unroll
  for (int e = 0; e < 8; ++e)
    b[8 + e] = (__bf16)base[n * stride_n + (k_base + 16 + koff + e) * stride_k];
  return b;
}

// Hot-path A loader: contiguous row-major f32, vectorized float4 (global_load_b128).
__device__ __forceinline__ v16bf load_a_rowmajor(const float* __restrict__ base, int ld,
                                                 int row_base, int k_base, int lane) {
  const int row  = row_base + (lane & 15);
  const int koff = ((lane >> 4) & 1) * 8;
  const float* p = base + (size_t)row * ld + k_base + koff;
  float4 x0 = *(const float4*)(p + 0);
  float4 x1 = *(const float4*)(p + 4);
  float4 x2 = *(const float4*)(p + 16);
  float4 x3 = *(const float4*)(p + 20);
  v16bf a;
  a[0]  = (__bf16)x0.x; a[1]  = (__bf16)x0.y; a[2]  = (__bf16)x0.z; a[3]  = (__bf16)x0.w;
  a[4]  = (__bf16)x1.x; a[5]  = (__bf16)x1.y; a[6]  = (__bf16)x1.z; a[7]  = (__bf16)x1.w;
  a[8]  = (__bf16)x2.x; a[9]  = (__bf16)x2.y; a[10] = (__bf16)x2.z; a[11] = (__bf16)x2.w;
  a[12] = (__bf16)x3.x; a[13] = (__bf16)x3.y; a[14] = (__bf16)x3.z; a[15] = (__bf16)x3.w;
  return a;
}

// ---------------------------------------------------------------------------
// Kernel 1: per batch (256 blocks, 1 wave each) compute
//   Q = q[a,b]Wq^T+bq ; K' = k[b,a]Wk^T+bk ; S = scale * Q^T K'^T ;
//   P = rowsoftmax(S) ; M = P Wv ; c = P bv        -> workspace
// ---------------------------------------------------------------------------
__global__ __launch_bounds__(32)
void attn_prep_kernel(const float* __restrict__ q, const float* __restrict__ k,
                      const float* __restrict__ Wq, const float* __restrict__ bq,
                      const float* __restrict__ Wk, const float* __restrict__ bk,
                      const float* __restrict__ Wv, const float* __restrict__ bv,
                      float* __restrict__ Mws, float* __restrict__ cws) {
  __shared__ float Qs[64 * 64];
  __shared__ float Ks[64 * 64];
  __shared__ float Ss[64 * 64];

  const int lane  = threadIdx.x;
  const int batch = blockIdx.x;          // batch = a*16 + b
  const int a_idx = batch >> 4;
  const int b_idx = batch & 15;
  const float* qb = q + (size_t)batch * 4096;                 // q[a,b]
  const float* kb = k + (size_t)(b_idx * 16 + a_idx) * 4096;  // k[b,a]  (swapaxes!)

  const int col   = lane & 15;
  const int half8 = ((lane >> 4) & 1) * 8;

  // ---- Q and K' projections: X[m,i] = sum_d x[m,d]*W[i,d] + bias[i] ----
  for (int mt = 0; mt < 4; ++mt) {
    v16bf Aq0 = load_a_frag(qb, 64, 1, mt * 16, 0,  lane);
    v16bf Aq1 = load_a_frag(qb, 64, 1, mt * 16, 32, lane);
    v16bf Ak0 = load_a_frag(kb, 64, 1, mt * 16, 0,  lane);
    v16bf Ak1 = load_a_frag(kb, 64, 1, mt * 16, 32, lane);
    for (int it = 0; it < 4; ++it) {
      // B[k=d, n=i] = W[i*64+d]  -> stride_n=64, stride_k=1
      v16bf Bq0 = load_b_frag(Wq, 64, 1, it * 16, 0,  lane);
      v16bf Bq1 = load_b_frag(Wq, 64, 1, it * 16, 32, lane);
      v8f accq = {};
      accq = wmma_bf16(Aq0, Bq0, accq);
      accq = wmma_bf16(Aq1, Bq1, accq);
      const float biasq = bq[it * 16 + col];
#pragma unroll
      for (int r = 0; r < 8; ++r)
        Qs[(mt * 16 + r + half8) * 64 + it * 16 + col] = accq[r] + biasq;

      v16bf Bk0 = load_b_frag(Wk, 64, 1, it * 16, 0,  lane);
      v16bf Bk1 = load_b_frag(Wk, 64, 1, it * 16, 32, lane);
      v8f acck = {};
      acck = wmma_bf16(Ak0, Bk0, acck);
      acck = wmma_bf16(Ak1, Bk1, acck);
      const float biask = bk[it * 16 + col];
#pragma unroll
      for (int r = 0; r < 8; ++r)
        Ks[(mt * 16 + r + half8) * 64 + it * 16 + col] = acck[r] + biask;
    }
  }
  __syncthreads();

  // ---- S[i,j] = scale * sum_m Qs[m][i] * Ks[j][m] ----
  for (int it = 0; it < 4; ++it) {
    // A[i,m] = Qs[m*64+i] -> stride_row=1, stride_k=64
    v16bf A0 = load_a_frag(Qs, 1, 64, it * 16, 0,  lane);
    v16bf A1 = load_a_frag(Qs, 1, 64, it * 16, 32, lane);
    for (int jt = 0; jt < 4; ++jt) {
      // B[k=m, n=j] = Ks[j*64+m] -> stride_n=64, stride_k=1
      v16bf B0 = load_b_frag(Ks, 64, 1, jt * 16, 0,  lane);
      v16bf B1 = load_b_frag(Ks, 64, 1, jt * 16, 32, lane);
      v8f acc = {};
      acc = wmma_bf16(A0, B0, acc);
      acc = wmma_bf16(A1, B1, acc);
#pragma unroll
      for (int r = 0; r < 8; ++r)
        Ss[(it * 16 + r + half8) * 64 + jt * 16 + col] = acc[r] * SCALE_QK;
    }
  }
  __syncthreads();

  // ---- row softmax over j: each lane owns rows lane and lane+32 ----
  for (int rr = 0; rr < 2; ++rr) {
    const int i = lane + rr * 32;
    float mx = -3.402823466e38f;
    for (int j = 0; j < 64; ++j) mx = fmaxf(mx, Ss[i * 64 + j]);
    float sum = 0.f;
    for (int j = 0; j < 64; ++j) {
      float e = __expf(Ss[i * 64 + j] - mx);
      Ss[i * 64 + j] = e;
      sum += e;
    }
    const float inv = 1.f / sum;
    for (int j = 0; j < 64; ++j) Ss[i * 64 + j] *= inv;
  }
  __syncthreads();

  // ---- M = P @ Wv ; c = P @ bv ----
  float* Mb = Mws + (size_t)batch * 4096;
  for (int it = 0; it < 4; ++it) {
    v16bf A0 = load_a_frag(Ss, 64, 1, it * 16, 0,  lane);
    v16bf A1 = load_a_frag(Ss, 64, 1, it * 16, 32, lane);
    for (int dt = 0; dt < 4; ++dt) {
      // B[k=j, n=d] = Wv[j*64+d] -> stride_n=1, stride_k=64
      v16bf B0 = load_b_frag(Wv, 1, 64, dt * 16, 0,  lane);
      v16bf B1 = load_b_frag(Wv, 1, 64, dt * 16, 32, lane);
      v8f acc = {};
      acc = wmma_bf16(A0, B0, acc);
      acc = wmma_bf16(A1, B1, acc);
#pragma unroll
      for (int r = 0; r < 8; ++r)
        Mb[(it * 16 + r + half8) * 64 + dt * 16 + col] = acc[r];
    }
  }
  for (int rr = 0; rr < 2; ++rr) {
    const int i = lane + rr * 32;
    float s = 0.f;
    for (int j = 0; j < 64; ++j) s += Ss[i * 64 + j] * bv[j];
    cws[batch * 64 + i] = s;
  }
}

// ---------------------------------------------------------------------------
// Kernel 2: out[a,b,t,i] = sum_d v[a,b,t,d] * M[i,d] + c[i]
// Memory-bound streaming GEMM: v (128 MiB) in, out (128 MiB) out.
// grid (4, 256), block 256 (8 waves). Each wave: 4 strips of 16 rows x 64 cols.
// ---------------------------------------------------------------------------
__global__ __launch_bounds__(256)
void attn_apply_kernel(const float* __restrict__ vsrc, const float* __restrict__ Mws,
                       const float* __restrict__ cws, float* __restrict__ out) {
  const int lane  = threadIdx.x & 31;
  const int wave  = threadIdx.x >> 5;
  const int batch = blockIdx.y;

  const float* vb = vsrc + (size_t)batch * 2048 * 64;
  const float* Mb = Mws + (size_t)batch * 4096;
  const float* cb = cws + (size_t)batch * 64;
  float* ob       = out + (size_t)batch * 2048 * 64;

  const int col   = lane & 15;
  const int half8 = ((lane >> 4) & 1) * 8;

  // Preload B = M^T fragments once per wave (stay in VGPRs).
  // B[k=d, n=i] = M[i*64+d] -> stride_n=64, stride_k=1 (rows of M are contiguous).
  v16bf Bf[4][2];
  float cval[4];
#pragma unroll
  for (int nt = 0; nt < 4; ++nt) {
    Bf[nt][0] = load_b_frag(Mb, 64, 1, nt * 16, 0,  lane);
    Bf[nt][1] = load_b_frag(Mb, 64, 1, nt * 16, 32, lane);
    cval[nt]  = cb[nt * 16 + col];
  }

#pragma unroll
  for (int iter = 0; iter < 4; ++iter) {
    const int strip = iter * 32 + blockIdx.x * 8 + wave;   // 0..127
    const int row0  = strip * 16;
    v16bf A0 = load_a_rowmajor(vb, 64, row0, 0,  lane);
    v16bf A1 = load_a_rowmajor(vb, 64, row0, 32, lane);
#pragma unroll
    for (int nt = 0; nt < 4; ++nt) {
      v8f acc = {};
      acc = wmma_bf16(A0, Bf[nt][0], acc);
      acc = wmma_bf16(A1, Bf[nt][1], acc);
#pragma unroll
      for (int r = 0; r < 8; ++r)
        ob[(size_t)(row0 + r + half8) * 64 + nt * 16 + col] = acc[r] + cval[nt];
    }
  }
}

extern "C" void kernel_launch(void* const* d_in, const int* in_sizes, int n_in,
                              void* d_out, int out_size, void* d_ws, size_t ws_size,
                              hipStream_t stream) {
  const float* q  = (const float*)d_in[0];
  const float* k  = (const float*)d_in[1];
  const float* v  = (const float*)d_in[2];
  const float* Wq = (const float*)d_in[3];
  const float* bq = (const float*)d_in[4];
  const float* Wk = (const float*)d_in[5];
  const float* bk = (const float*)d_in[6];
  const float* Wv = (const float*)d_in[7];
  const float* bv = (const float*)d_in[8];
  float* out = (float*)d_out;

  // Workspace: M (256 x 64 x 64 f32 = 4 MiB) then c (256 x 64 f32 = 64 KiB)
  float* Mws = (float*)d_ws;
  float* cws = Mws + 256 * 64 * 64;

  attn_prep_kernel<<<dim3(256), dim3(32), 0, stream>>>(q, k, Wq, bq, Wk, bk, Wv, bv,
                                                       Mws, cws);
  attn_apply_kernel<<<dim3(4, 256), dim3(256), 0, stream>>>(v, Mws, cws, out);
}
